// AdjConstructor_75153337745450
// MI455X (gfx1250) — matrix operand order repaired and checked by reading, not compiled, not run on hardware
//
#include <hip/hip_runtime.h>
#include <hip/hip_bf16.h>

#define N_NODES 8192
#define EMB_DIM 128
#define ALPHA   3.0f
#define TOPK    16
#define TPB     256

typedef _Float16 v16h __attribute__((ext_vector_type(16)));
typedef _Float16 v8h  __attribute__((ext_vector_type(8)));
typedef float    v8f  __attribute__((ext_vector_type(8)));

// Hardware tanh (v_tanh_f32) on gfx1250 if the builtin exists; OCML otherwise.
static __device__ inline float hw_tanh(float x) {
#if __has_builtin(__builtin_amdgcn_tanhf)
    return __builtin_amdgcn_tanhf(x);
#else
    return tanhf(x);
#endif
}

// ---------------------------------------------------------------------------
// Stage 1: e = tanh(ALPHA * (emb[idx] @ W^T + b)), stored as f16 [N, D].
// One block per row; 128 threads; x-row staged in LDS, W rows stream from L2.
// ---------------------------------------------------------------------------
__global__ __launch_bounds__(EMB_DIM)
void embed_tanh_f16(const int* __restrict__ idx,
                    const float* __restrict__ emb,
                    const float* __restrict__ W,
                    const float* __restrict__ b,
                    _Float16* __restrict__ out) {
    __shared__ float xrow[EMB_DIM];
    const int r = blockIdx.x;
    const int c = threadIdx.x;
    const int src = idx[r];
    xrow[c] = emb[(size_t)src * EMB_DIM + c];
    __syncthreads();

    const float* wrow = W + (size_t)c * EMB_DIM;  // out[c] = sum_k x[k]*W[c,k]
    float acc = b[c];
#pragma unroll 8
    for (int k = 0; k < EMB_DIM; ++k) {
        acc = fmaf(xrow[k], wrow[k], acc);
    }
    out[(size_t)r * EMB_DIM + c] = (_Float16)hw_tanh(ALPHA * acc);
}

// ---------------------------------------------------------------------------
// WMMA fragment loaders (wave32, V_WMMA_F32_16X16X32_F16; contiguous k order, same in A and B).
//
// A (16x32 MxK, 16-bit): lane L holds row M = L&15.
//   lanes 0-15 : halves 0..15 = K[kk+0..15]
//   lanes 16-31: halves 0..15 = K[kk+16..31]
// rowp points at the lane's matrix row; two 16B loads + shuffle.
// ---------------------------------------------------------------------------
static __device__ inline v16h load_a_frag(const _Float16* rowp, int kk, int hi) {
    const _Float16* p = rowp + kk + hi * 16;
    v8h lo = *(const v8h*)(p);        // K base .. base+7
    v8h up = *(const v8h*)(p + 8);    // K base+8 .. base+15
    return __builtin_shufflevector(lo, up,
        0, 1, 2, 3, 4, 5, 6, 7, 8, 9, 10, 11, 12, 13, 14, 15);
}

// B (32x16 KxN, 16-bit): lane L holds column N = L&15.
//   lanes 0-15 : halves = B[kk+0..15][N]   lanes 16-31: halves = B[kk+16..31][N]
// Here B[k][n] = e[j0+n][k], so a lane's fragment is 32 contiguous bytes of an
// e-row. One aligned 32B load.
static __device__ inline v16h load_b_frag(const _Float16* rowp, int kk, int hi) {
    return *(const v16h*)(rowp + kk + hi * 16);
}

// ---------------------------------------------------------------------------
// Stage 2: adj = relu(tanh(ALPHA*(e1@e2^T - e2@e1^T))), f32 [N, N].
// 8 waves / block; each wave computes one 32x32 macro-tile = 2x2 WMMA tiles,
// two accumulators each (antisymmetric pair) -> 32 v_wmma per wave, with A/B
// fragments reused across the 2x2 tiles (halves L2 fragment traffic).
// grid = (N/32, N/32/8) = (256, 32).
// ---------------------------------------------------------------------------
__global__ __launch_bounds__(TPB)
void adj_wmma(const _Float16* __restrict__ e1,
              const _Float16* __restrict__ e2,
              float* __restrict__ out) {
    const int lane = threadIdx.x & 31;
    const int wave = threadIdx.x >> 5;
    const int i0 = (blockIdx.y * 8 + wave) * 32;  // macro-tile row base
    const int j0 = blockIdx.x * 32;               // macro-tile col base
    const int m  = lane & 15;
    const int hi = lane >> 4;                     // which half-wave

    // A-row pointers (2 i-tiles x {e1,e2}) and B-col pointers (2 j-tiles)
    const _Float16* a1p[2] = { e1 + (size_t)(i0 + m) * EMB_DIM,
                               e1 + (size_t)(i0 + 16 + m) * EMB_DIM };
    const _Float16* a2p[2] = { e2 + (size_t)(i0 + m) * EMB_DIM,
                               e2 + (size_t)(i0 + 16 + m) * EMB_DIM };
    const _Float16* b2p[2] = { e2 + (size_t)(j0 + m) * EMB_DIM,
                               e2 + (size_t)(j0 + 16 + m) * EMB_DIM };
    const _Float16* b1p[2] = { e1 + (size_t)(j0 + m) * EMB_DIM,
                               e1 + (size_t)(j0 + 16 + m) * EMB_DIM };

    v8f acc1[2][2] = {};  // e1_i . e2_j
    v8f acc2[2][2] = {};  // e2_i . e1_j

#pragma unroll
    for (int kk = 0; kk < EMB_DIM; kk += 32) {
        v16h A1[2], A2[2], B1[2], B2[2];
#pragma unroll
        for (int t = 0; t < 2; ++t) {
            A1[t] = load_a_frag(a1p[t], kk, hi);
            A2[t] = load_a_frag(a2p[t], kk, hi);
            B2[t] = load_b_frag(b2p[t], kk, hi);
            B1[t] = load_b_frag(b1p[t], kk, hi);
        }
#pragma unroll
        for (int ii = 0; ii < 2; ++ii) {
#pragma unroll
            for (int jj = 0; jj < 2; ++jj) {
                acc1[ii][jj] = __builtin_amdgcn_wmma_f32_16x16x32_f16(
                    false, A1[ii], false, B2[jj], (short)0, acc1[ii][jj], false, false);
                acc2[ii][jj] = __builtin_amdgcn_wmma_f32_16x16x32_f16(
                    false, A2[ii], false, B1[jj], (short)0, acc2[ii][jj], false, false);
            }
        }
    }

    // C/D layout: VGPR r -> row +r (+8 for lanes 16-31), col = +(lane&15)
#pragma unroll
    for (int ii = 0; ii < 2; ++ii) {
#pragma unroll
        for (int jj = 0; jj < 2; ++jj) {
            const int row0 = i0 + ii * 16 + hi * 8;
            const int col  = j0 + jj * 16 + m;
#pragma unroll
            for (int r = 0; r < 8; ++r) {
                float t = hw_tanh(ALPHA * (acc1[ii][jj][r] - acc2[ii][jj][r]));
                out[(size_t)(row0 + r) * N_NODES + col] = t > 0.0f ? t : 0.0f;
            }
        }
    }
}

// ---------------------------------------------------------------------------
// Stage 3: per-row top-16 with lowest-index tie-break; zero everything else.
// One block per row. Each thread keeps a sorted top-16 (key = (value, -idx))
// of its strided slice in LDS, tree-merge to the row's top-16, then rewrite
// the row keeping only keys >= the 16th key.
// ---------------------------------------------------------------------------
__global__ __launch_bounds__(TPB)
void topk_mask(float* __restrict__ adj) {
    __shared__ float sv[TPB * TOPK];
    __shared__ int   si[TPB * TOPK];
    __shared__ float s_vstar;
    __shared__ int   s_istar;

    const int tid = threadIdx.x;
    float* row = adj + (size_t)blockIdx.x * N_NODES;

    float* mv = &sv[tid * TOPK];
    int*   mi = &si[tid * TOPK];
#pragma unroll
    for (int t = 0; t < TOPK; ++t) { mv[t] = -1.0f; mi[t] = 0x7fffffff; }

    // local top-16 of this thread's strided slice (insertion sort, desc by key)
    for (int j = tid; j < N_NODES; j += TPB) {
        float v = row[j];
        float lv = mv[TOPK - 1]; int li = mi[TOPK - 1];
        if (v > lv || (v == lv && j < li)) {
            int p = TOPK - 1;
            while (p > 0) {
                float pv = mv[p - 1]; int pi = mi[p - 1];
                if (v > pv || (v == pv && j < pi)) {
                    mv[p] = pv; mi[p] = pi; --p;
                } else break;
            }
            mv[p] = v; mi[p] = j;
        }
    }
    __syncthreads();

    // tree merge of sorted-16 lists
    for (int s = TPB / 2; s >= 1; s >>= 1) {
        if (tid < s) {
            const float* av = &sv[tid * TOPK];
            const int*   ai = &si[tid * TOPK];
            const float* bv = &sv[(tid + s) * TOPK];
            const int*   bi = &si[(tid + s) * TOPK];
            int ia = 0, ib = 0;
            float rv[TOPK]; int ri[TOPK];
#pragma unroll
            for (int t = 0; t < TOPK; ++t) {
                float va = av[ia]; int ja = ai[ia];
                float vb = bv[ib]; int jb = bi[ib];
                bool takeA = (va > vb) || (va == vb && ja < jb);
                rv[t] = takeA ? va : vb;
                ri[t] = takeA ? ja : jb;
                if (takeA) ++ia; else ++ib;
            }
#pragma unroll
            for (int t = 0; t < TOPK; ++t) { sv[tid * TOPK + t] = rv[t]; si[tid * TOPK + t] = ri[t]; }
        }
        __syncthreads();
    }

    if (tid == 0) { s_vstar = sv[TOPK - 1]; s_istar = si[TOPK - 1]; }
    __syncthreads();
    const float vstar = s_vstar;
    const int   istar = s_istar;

    // keep element j iff key(v,j) >= key(vstar,istar): exactly the top-16 keys
    for (int j = tid; j < N_NODES; j += TPB) {
        float v = row[j];
        bool keep = (v > vstar) || (v == vstar && j <= istar);
        row[j] = keep ? v : 0.0f;
    }
}

// ---------------------------------------------------------------------------
extern "C" void kernel_launch(void* const* d_in, const int* in_sizes, int n_in,
                              void* d_out, int out_size, void* d_ws, size_t ws_size,
                              hipStream_t stream) {
    const int*   idx  = (const int*)d_in[0];
    const float* emb1 = (const float*)d_in[1];
    const float* emb2 = (const float*)d_in[2];
    const float* th1w = (const float*)d_in[3];
    const float* th1b = (const float*)d_in[4];
    const float* th2w = (const float*)d_in[5];
    const float* th2b = (const float*)d_in[6];
    float* out = (float*)d_out;

    _Float16* e1h = (_Float16*)d_ws;                        // 2 MB
    _Float16* e2h = e1h + (size_t)N_NODES * EMB_DIM;        // 2 MB

    embed_tanh_f16<<<N_NODES, EMB_DIM, 0, stream>>>(idx, emb1, th1w, th1b, e1h);
    embed_tanh_f16<<<N_NODES, EMB_DIM, 0, stream>>>(idx, emb2, th2w, th2b, e2h);

    dim3 grid(N_NODES / 32, N_NODES / 32 / 8);   // (256, 32), 8 waves/block
    adj_wmma<<<grid, TPB, 0, stream>>>(e1h, e2h, out);

    topk_mask<<<N_NODES, TPB, 0, stream>>>(out);
}
